// DeformConv2d_7490422964780
// MI455X (gfx1250) — matrix operand order repaired and checked by reading, not compiled, not run on hardware
//
#include <hip/hip_runtime.h>
#include <hip/hip_bf16.h>

typedef __attribute__((ext_vector_type(16))) _Float16 v16h;
typedef __attribute__((ext_vector_type(8)))  _Float16 v8h;
typedef __attribute__((ext_vector_type(8)))  float    v8f;

#define Cc   256
#define Oc   256
#define Hh   64
#define Ww   64
#define HW   4096
#define K2   9
#define KD   2304          // C*K2 = total GEMM K; 72 chunks of 32
#define NCH  72
#define BROW 40            // B-tile LDS row stride in f16 (80 B, padded from 64)

union AFrag { v16h v; v8h h[2]; };

// ---------------------------------------------------------------------------
// Prepass: repack weight f32 [O][C][3][3] -> f16 Wt[O][kk*256 + c]
// K-dim ordered (kk outer, c inner): chunk q's 32 columns are Wt row + q*32.
// ---------------------------------------------------------------------------
__global__ __launch_bounds__(256) void pack_weights(const float* __restrict__ w,
                                                    _Float16* __restrict__ wt) {
    int id = blockIdx.x * 256 + threadIdx.x;
    if (id >= Oc * KD) return;
    int o  = id / KD;
    int r  = id % KD;
    int c  = r / K2;
    int kk = r % K2;
    wt[o * KD + kk * Cc + c] = (_Float16)w[id];
}

// ---------------------------------------------------------------------------
// Fused deformable-conv with software-pipelined (double-buffered) LDS tiles:
// gathers for chunk q+1 overlap the 8 WMMAs of chunk q.
// One workgroup: batch b, one image row h (64 px), all 256 output channels.
// ---------------------------------------------------------------------------
__global__ __launch_bounds__(256) void deform_conv_wmma(
    const float* __restrict__ inp,     // [B][C][64][64]
    const float* __restrict__ offset,  // [B][18][64][64]
    const float* __restrict__ mask,    // [B][9][64][64]
    const _Float16* __restrict__ wt,   // [O][KD] packed
    const float* __restrict__ bias,    // [O]
    float* __restrict__ out)           // [B][O][64][64]
{
    __shared__ _Float16 BlA[64 * BROW];   // ping
    __shared__ _Float16 BlB[64 * BROW];   // pong
    __shared__ int   sidx[64 * 4];        // clamped corner indices per pixel
    __shared__ float swgt[64 * 4];        // bilinear*mask weights per pixel

    const int t       = threadIdx.x;
    const int lane    = t & 31;
    const int wave    = t >> 5;
    const int lane_lo = lane & 15;
    const int lane_hi = lane >> 4;

    const int h = blockIdx.x;             // image row == 64-pixel N tile
    const int b = blockIdx.y;

    const float* inp_b  = inp    + (size_t)b * Cc * HW;
    const float* off_b  = offset + (size_t)b * 18 * HW + h * Ww;
    const float* mask_b = mask   + (size_t)b * K2 * HW + h * Ww;

    const int cp = t >> 2;                // coord phase: pixel
    const int cj = t & 3;                 // coord phase: corner
    const int myp  = t & 63;              // build phase: pixel (lanes->pixels)
    const int quad = t >> 6;              // build phase: channel sub-group

    int4   i4;
    float4 w4;

    // per-(kk,pixel) bilinear coordinates -> LDS
    auto coords = [&](int kk) {
        float oy = off_b [(kk * 2 + 0) * HW + cp];
        float ox = off_b [(kk * 2 + 1) * HW + cp];
        float mk = mask_b[ kk          * HW + cp];
        float y = (float)(h  - 1 + kk / 3) + oy;
        float x = (float)(cp - 1 + kk % 3) + ox;
        float y0f = floorf(y), x0f = floorf(x);
        float ly = y - y0f, lx = x - x0f;
        float hy = 1.f - ly, hx = 1.f - lx;
        int cy = (int)y0f + (cj >> 1);
        int cx = (int)x0f + (cj & 1);
        float wj = ((cj >> 1) ? ly : hy) * ((cj & 1) ? lx : hx) * mk;
        bool valid = (cy >= 0) & (cy < Hh) & (cx >= 0) & (cx < Ww);
        if (!valid) wj = 0.f;
        int icy = min(max(cy, 0), Hh - 1);
        int icx = min(max(cx, 0), Ww - 1);
        sidx[cp * 4 + cj] = icy * Ww + icx;
        swgt[cp * 4 + cj] = wj;
    };
    auto load_coord_regs = [&]() {
        i4 = *(const int4*)  (&sidx[myp * 4]);
        w4 = *(const float4*)(&swgt[myp * 4]);
    };

    v8f acc[2][4];
    #pragma unroll
    for (int m = 0; m < 2; ++m)
        #pragma unroll
        for (int n = 0; n < 4; ++n)
            acc[m][n] = (v8f)0.f;

    // one pipeline stage: WMMA chunk q from rbuf; gather chunk q+1 into wbuf
    auto step = [&](int q, const _Float16* rbuf, _Float16* wbuf) {
        __syncthreads();                          // rbuf fully written
        const int nq = q + 1;
        const bool do_build = (nq < NCH);
        float vals[8];
        if (do_build) {
            if ((nq & 7) == 0) {                  // kernel-tap boundary
                coords(nq >> 3);
                __syncthreads();
                load_coord_regs();
            }
            const int ncb = (nq & 7) * 32;
            #pragma unroll
            for (int i = 0; i < 8; ++i) {         // 32 gathers, in flight
                const float* plane = inp_b + (size_t)(ncb + quad + 4 * i) * HW;
                vals[i] = w4.x * plane[i4.x] + w4.y * plane[i4.y]
                        + w4.z * plane[i4.z] + w4.w * plane[i4.w];
            }
        }
        // A fragments: ISA 16-bit A 16x32 layout (two 16B chunks per lane)
        AFrag a[2];
        #pragma unroll
        for (int m = 0; m < 2; ++m) {
            const _Float16* arow =
                wt + (size_t)(wave * 32 + m * 16 + lane_lo) * KD + q * 32;
            a[m].h[0] = *(const v8h*)(arow +      lane_hi * 8);
            a[m].h[1] = *(const v8h*)(arow + 16 + lane_hi * 8);
        }
        // B fragments: same K layout as A (lane_hi*8 and 16 + lane_hi*8), col = lane_lo
        AFrag bf[4];
        #pragma unroll
        for (int n = 0; n < 4; ++n) {
            const _Float16* brow = &rbuf[(n * 16 + lane_lo) * BROW + lane_hi * 8];
            bf[n].h[0] = *(const v8h*)(brow);
            bf[n].h[1] = *(const v8h*)(brow + 16);
        }
        #pragma unroll
        for (int m = 0; m < 2; ++m)
            #pragma unroll
            for (int n = 0; n < 4; ++n)
                acc[m][n] = __builtin_amdgcn_wmma_f32_16x16x32_f16(
                    false, a[m].v, false, bf[n].v,
                    (short)0, acc[m][n], false, false);
        if (do_build) {
            #pragma unroll
            for (int i = 0; i < 8; ++i)           // commit after WMMAs
                wbuf[myp * BROW + quad + 4 * i] = (_Float16)vals[i];
        }
    };

    // prologue: coords(kk=0) and build chunk 0 into ping buffer
    coords(0);
    __syncthreads();
    load_coord_regs();
    #pragma unroll
    for (int i = 0; i < 8; ++i) {
        const float* plane = inp_b + (size_t)(quad + 4 * i) * HW;
        float v = w4.x * plane[i4.x] + w4.y * plane[i4.y]
                + w4.z * plane[i4.z] + w4.w * plane[i4.w];
        BlA[myp * BROW + quad + 4 * i] = (_Float16)v;
    }

    // main pipeline: 72 chunks, ping-pong with compile-time buffer pointers
    for (int q = 0; q < NCH; q += 2) {
        step(q,     BlA, BlB);
        step(q + 1, BlB, BlA);
    }

    // epilogue: bias + store (C/D layout: VGPR j -> M = j + 8*lane_hi)
    #pragma unroll
    for (int m = 0; m < 2; ++m) {
        #pragma unroll
        for (int j = 0; j < 8; ++j) {
            const int o = wave * 32 + m * 16 + j + 8 * lane_hi;
            const float bv = bias[o];
            float* orow = out + ((size_t)b * Oc + o) * HW + h * Ww + lane_lo;
            #pragma unroll
            for (int n = 0; n < 4; ++n)
                orow[n * 16] = acc[m][n][j] + bv;
        }
    }
}

// ---------------------------------------------------------------------------
extern "C" void kernel_launch(void* const* d_in, const int* in_sizes, int n_in,
                              void* d_out, int out_size, void* d_ws, size_t ws_size,
                              hipStream_t stream) {
    const float* inp    = (const float*)d_in[0];
    const float* offset = (const float*)d_in[1];
    const float* mask   = (const float*)d_in[2];
    const float* weight = (const float*)d_in[3];
    const float* bias   = (const float*)d_in[4];
    float* out = (float*)d_out;

    _Float16* wt = (_Float16*)d_ws;   // 256*2304*2 = 1.18 MB packed weights

    pack_weights<<<(Oc * KD + 255) / 256, 256, 0, stream>>>(weight, wt);

    dim3 grid(Hh, 8, 1);              // 64 rows x 8 batches = 512 workgroups
    deform_conv_wmma<<<grid, 256, 0, stream>>>(inp, offset, mask, wt, bias, out);
}
